// MultiHeadAttention_61967788146794
// MI455X (gfx1250) — matrix operand order repaired and checked by reading, not compiled, not run on hardware
//
#include <hip/hip_runtime.h>

#define S_LEN 4096
#define D_LEN 768
#define NH    12
#define DHEAD 64
#define QT    (S_LEN / 16)   // 256 query tiles
#define KSUBD (D_LEN / 32)   // 24 k-subtiles for D-wide GEMMs
#define NTD   (D_LEN / 16)   // 48 n-tiles for D-wide GEMMs

typedef __attribute__((ext_vector_type(16))) __bf16          v16bf;
typedef __attribute__((ext_vector_type(16))) unsigned short  v16u;
typedef __attribute__((ext_vector_type(8)))  unsigned short  v8u;
typedef __attribute__((ext_vector_type(8)))  float           v8f;

// ---- fp32 -> bf16 bits, round-to-nearest-even ----
static __device__ __forceinline__ unsigned short f2bf(float f) {
    union { float f; unsigned int u; } v; v.f = f;
    unsigned int x = v.u;
    unsigned int r = x + 0x7FFFu + ((x >> 16) & 1u);
    return (unsigned short)(r >> 16);
}

// ---- bf16 bits -> fp32 (one shift) ----
static __device__ __forceinline__ float bf2f(unsigned short u) {
    union { unsigned int u; float f; } v;
    v.u = ((unsigned int)u) << 16;
    return v.f;
}

static __device__ __forceinline__ v16bf as_bf(v16u u) {
    return __builtin_bit_cast(v16bf, u);
}

static __device__ __forceinline__ v8f wmma_bf16(v16bf a, v16bf b, v8f c) {
    // 8 args: (neg_a, A, neg_b, B, c_mod, C, reuse_a, reuse_b)
    return __builtin_amdgcn_wmma_f32_16x16x32_bf16(false, a, false, b, (short)0, c, false, false);
}

// ---- workspace layout (units: unsigned short / bf16 elements) ----
#define W_SZ    ((size_t)(KSUBD * NTD * 512))          // 589824  per weight
#define IN_SZ   ((size_t)(QT * KSUBD * 512))           // 3145728 per input
#define OFF_W   ((size_t)0)                            // 4 weights
#define OFF_IN  (OFF_W + 4 * W_SZ)                     // 3 inputs (A-frags)
#define OFF_PQ  (OFF_IN + 3 * IN_SZ)                   // pQ A-frags: [H][QT][2][512]
#define OFF_PK  (OFF_PQ + IN_SZ)                       // pK B-frags: [H][QT][2][512]
#define OFF_PV  (OFF_PK + IN_SZ)                       // pV B-frags: [H][128][4][512]
#define OFF_CTX (OFF_PV + IN_SZ)                       // context A-frags: [QT][24][512]

// =====================================================================
// K1: pack weights -> bf16 B-fragments.  B[k][n] = W[n][k]  (folds W^T)
// B-frag layout (32x16 bf16): lane = (n&15) + 16*h ; half j -> K = 8*h + (j&7) + (j>=8 ? 16 : 0)
// =====================================================================
__global__ void pack_w_kernel(const float* __restrict__ Wq, const float* __restrict__ Wk,
                              const float* __restrict__ Wv, const float* __restrict__ Wo,
                              unsigned short* __restrict__ ws) {
    int tid  = blockIdx.x * blockDim.x + threadIdx.x;   // < 4*24*48*32
    int lane = tid & 31;
    int t    = tid >> 5;
    int ntile = t % NTD;  t /= NTD;
    int ksub  = t % KSUBD; t /= KSUBD;
    int w     = t;                                      // 0..3
    const float* Wsrc = (w == 0) ? Wq : (w == 1) ? Wk : (w == 2) ? Wv : Wo;
    int n = lane & 15, h = lane >> 4;
    const float* src = Wsrc + (size_t)(ntile * 16 + n) * D_LEN + ksub * 32 + 8 * h;
    float4 f0 = ((const float4*)src)[0];
    float4 f1 = ((const float4*)src)[1];
    float4 f2 = ((const float4*)(src + 16))[0];
    float4 f3 = ((const float4*)(src + 16))[1];
    v16u o;
    o[0]=f2bf(f0.x);  o[1]=f2bf(f0.y);  o[2]=f2bf(f0.z);  o[3]=f2bf(f0.w);
    o[4]=f2bf(f1.x);  o[5]=f2bf(f1.y);  o[6]=f2bf(f1.z);  o[7]=f2bf(f1.w);
    o[8]=f2bf(f2.x);  o[9]=f2bf(f2.y);  o[10]=f2bf(f2.z); o[11]=f2bf(f2.w);
    o[12]=f2bf(f3.x); o[13]=f2bf(f3.y); o[14]=f2bf(f3.z); o[15]=f2bf(f3.w);
    unsigned short* dst = ws + OFF_W + (size_t)w * W_SZ
                          + ((size_t)(ksub * NTD + ntile)) * 512 + lane * 16;
    *(v16u*)dst = o;
}

// =====================================================================
// K2: pack activations Q/K/V -> bf16 A-fragments.
// A-frag layout (16x32 bf16, ISA 7.12.2): lane = m + 16*h ;
//   half j -> K = 8*h + (j&7) + (j>=8 ? 16 : 0)
// =====================================================================
__global__ void pack_x_kernel(const float* __restrict__ Q, const float* __restrict__ K,
                              const float* __restrict__ V,
                              unsigned short* __restrict__ ws) {
    int tid  = blockIdx.x * blockDim.x + threadIdx.x;   // < 3*256*24*32
    int lane = tid & 31;
    int t    = tid >> 5;
    int ksub  = t % KSUBD; t /= KSUBD;
    int qtile = t % QT;    t /= QT;
    int z     = t;                                      // 0..2
    const float* X = (z == 0) ? Q : (z == 1) ? K : V;
    int m = lane & 15, h = lane >> 4;
    const float* src = X + (size_t)(qtile * 16 + m) * D_LEN + ksub * 32 + 8 * h;
    float4 f0 = ((const float4*)src)[0];        // K = 8h+0..3
    float4 f1 = ((const float4*)src)[1];        // K = 8h+4..7
    float4 f2 = ((const float4*)(src + 16))[0]; // K = 8h+16..19
    float4 f3 = ((const float4*)(src + 16))[1]; // K = 8h+20..23
    v16u o;
    o[0]=f2bf(f0.x);  o[1]=f2bf(f0.y);  o[2]=f2bf(f0.z);  o[3]=f2bf(f0.w);
    o[4]=f2bf(f1.x);  o[5]=f2bf(f1.y);  o[6]=f2bf(f1.z);  o[7]=f2bf(f1.w);
    o[8]=f2bf(f2.x);  o[9]=f2bf(f2.y);  o[10]=f2bf(f2.z); o[11]=f2bf(f2.w);
    o[12]=f2bf(f3.x); o[13]=f2bf(f3.y); o[14]=f2bf(f3.z); o[15]=f2bf(f3.w);
    unsigned short* dst = ws + OFF_IN + (size_t)z * IN_SZ
                          + ((size_t)(qtile * KSUBD + ksub)) * 512 + lane * 16;
    *(v16u*)dst = o;
}

// =====================================================================
// K3: QKV projections: C = X @ W^T + b  via WMMA; epilogue scatters bf16
// results into the attention-stage fragment layouts.
// grid (256, 12, 3), block 128 (4 waves, each one 16-wide n-tile)
// =====================================================================
__global__ void proj_kernel(const unsigned short* ws_r, unsigned short* ws_w,
                            const float* __restrict__ bq, const float* __restrict__ bk,
                            const float* __restrict__ bv) {
    int z    = blockIdx.z;
    int qt   = blockIdx.x;
    int wid  = threadIdx.x >> 5;
    int lane = threadIdx.x & 31;
    int ntile = blockIdx.y * 4 + wid;                   // 0..47

    const unsigned short* A = ws_r + OFF_IN + (size_t)z * IN_SZ + (size_t)qt * KSUBD * 512;
    const unsigned short* B = ws_r + OFF_W + (size_t)z * W_SZ;

    v8f c = {};
    for (int k = 0; k < KSUBD; ++k) {
        v16bf a = *(const v16bf*)(A + (size_t)k * 512 + lane * 16);
        v16bf b = *(const v16bf*)(B + ((size_t)(k * NTD + ntile)) * 512 + lane * 16);
        c = wmma_bf16(a, b, c);
    }

    const float* bias = (z == 0) ? bq : (z == 1) ? bk : bv;
    int n = lane & 15, hh = lane >> 4;
    float bb = bias[ntile * 16 + n];
    int col  = ntile * 16 + n;
    int head = col >> 6;
    int dh   = col & 63;

    unsigned short* outQ = ws_w + OFF_PQ;
    unsigned short* outK = ws_w + OFF_PK;
    unsigned short* outV = ws_w + OFF_PV;

    for (int r = 0; r < 8; ++r) {
        int mm   = r + 8 * hh;                 // C layout: M = r + 8*(lane>=16)
        int rowg = qt * 16 + mm;
        unsigned short val = f2bf(c[r] + bb);
        if (z == 0) {           // pQ as A-frag over (q-tile, dh-subtile)
            int dhsub = dh >> 5, Kk = dh & 31;
            int h2 = (Kk >> 3) & 1;
            int j2 = (Kk & 7) + ((Kk >> 4) << 3);
            outQ[((size_t)((head * QT + qt) * 2 + dhsub)) * 512 + (mm + 16 * h2) * 16 + j2] = val;
        } else if (z == 1) {    // pK as B-frag: B[dh][key] = pK^T
            int dhsub = dh >> 5, Kk = dh & 31;
            outK[((size_t)((head * QT + qt) * 2 + dhsub)) * 512 + (mm + 16 * ((Kk >> 3) & 1)) * 16 + (Kk & 7) + ((Kk >> 4) << 3)] = val;
        } else {                // pV as B-frag: B[key][dh]
            int keysub = rowg >> 5, Kk = rowg & 31;
            int dht = dh >> 4;
            outV[((size_t)((head * 128 + keysub) * 4 + dht)) * 512 + ((dh & 15) + 16 * ((Kk >> 3) & 1)) * 16 + (Kk & 7) + ((Kk >> 4) << 3)] = val;
        }
    }
}

// =====================================================================
// K4: fused attention. LDS holds bf16 scores ALREADY IN A-FRAGMENT LAYOUT
//   sbf[ks 0..127][lane 0..31][j 0..15]  (128 KB),
//   frag (ks,lane,j) <-> (row = lane&15, key = ks*32 + 8*(lane>>4) + (j&7) + (j>=8?16:0))
// Phase 1: QK^T WMMA -> scatter bf16 scores into frags.
// Phase 2: softmax; exp kept UNNORMALIZED in frags (owner-exclusive slots,
//          no races); normalized fp32 probs streamed to HBM once (float4).
// Phase 3: PV WMMA with A-frags as direct v16bf LDS loads; 1/rowsum folded
//          into the fp32 epilogue.
// grid (256, 12), block 128, dynamic LDS = 128KB + 640B.
// =====================================================================
__global__ void attn_kernel(const unsigned short* ws, unsigned short* ctx_out,
                            float* __restrict__ attnOut) {
    extern __shared__ unsigned short sbf[];            // [128*512] bf16 frags
    float* red    = (float*)(sbf + 128 * 512);         // [128]
    float* rowmax = red + 128;                         // [16]
    float* rowsum = rowmax + 16;                       // [16]

    const int qt   = blockIdx.x;
    const int head = blockIdx.y;
    const int tid  = threadIdx.x;
    const int wid  = tid >> 5;
    const int lane = tid & 31;
    const int m    = lane & 15;
    const int hh   = lane >> 4;

    const unsigned short* pQ = ws + OFF_PQ;
    const unsigned short* pK = ws + OFF_PK;
    const unsigned short* pV = ws + OFF_PV;

    // ---- phase 1: scores = (pQ @ pK^T) / DH -> bf16 frags in LDS ----
    const unsigned short* aq = pQ + ((size_t)((head * QT + qt) * 2)) * 512;
    v16bf a0 = *(const v16bf*)(aq + lane * 16);
    v16bf a1 = *(const v16bf*)(aq + 512 + lane * 16);
    for (int kt = wid * 64; kt < wid * 64 + 64; ++kt) {
        const unsigned short* bk_ = pK + ((size_t)((head * QT + kt) * 2)) * 512;
        v16bf b0 = *(const v16bf*)(bk_ + lane * 16);
        v16bf b1 = *(const v16bf*)(bk_ + 512 + lane * 16);
        v8f c = {};
        c = wmma_bf16(a0, b0, c);
        c = wmma_bf16(a1, b1, c);
        int key = kt * 16 + m;                  // N = lane&15
        int ks  = key >> 5, Kk = key & 31;
        int h2  = (Kk >> 3) & 1;
        int j2  = (Kk & 7) + ((Kk >> 4) << 3);
        for (int r = 0; r < 8; ++r) {
            int row = r + 8 * hh;               // M = r + 8*(lane>=16)
            sbf[((ks * 32 + row + 16 * h2) << 4) + j2] = f2bf(c[r] * (1.0f / (float)DHEAD));
        }
    }
    __syncthreads();

    // ---- phase 2: softmax (8 threads per row; owner-exclusive frag slots) ----
    {
        int row = tid >> 3;                     // 0..15
        int sub = tid & 7;                      // 0..7  -> ks in [sub*16, sub*16+16)
        // pass A: row max
        float mx = -3.402823466e38f;
        for (int ks = sub * 16; ks < sub * 16 + 16; ++ks) {
            for (int h2 = 0; h2 < 2; ++h2) {
                unsigned short* p = sbf + ((ks * 32 + row + 16 * h2) << 4);
                v8u a = *(v8u*)p;
                v8u b = *(v8u*)(p + 8);
                for (int k = 0; k < 8; ++k) mx = fmaxf(mx, bf2f(a[k]));
                for (int k = 0; k < 8; ++k) mx = fmaxf(mx, bf2f(b[k]));
            }
        }
        red[tid] = mx;
        __syncthreads();
        if (sub == 0) {
            float v = red[tid];
            for (int i = 1; i < 8; ++i) v = fmaxf(v, red[tid + i]);
            rowmax[row] = v;
        }
        __syncthreads();
        mx = rowmax[row];
        // pass B: e = exp(x - mx) written back in place (bf16), fp32 row sum
        float sum = 0.0f;
        for (int ks = sub * 16; ks < sub * 16 + 16; ++ks) {
            for (int h2 = 0; h2 < 2; ++h2) {
                unsigned short* p = sbf + ((ks * 32 + row + 16 * h2) << 4);
                v8u a = *(v8u*)p;
                v8u b = *(v8u*)(p + 8);
                v8u ea, eb;
                for (int k = 0; k < 8; ++k) { float e = __expf(bf2f(a[k]) - mx); sum += e; ea[k] = f2bf(e); }
                for (int k = 0; k < 8; ++k) { float e = __expf(bf2f(b[k]) - mx); sum += e; eb[k] = f2bf(e); }
                *(v8u*)p = ea;
                *(v8u*)(p + 8) = eb;
            }
        }
        red[tid] = sum;
        __syncthreads();
        if (sub == 0) {
            float v = 0.0f;
            for (int i = 0; i < 8; ++i) v += red[tid + i];
            rowsum[row] = v;
        }
        __syncthreads();
        // pass C: stream normalized fp32 probs to HBM (float4 runs of 8 keys)
        float inv = 1.0f / rowsum[row];
        float* grow = attnOut + ((size_t)head * S_LEN + (size_t)(qt * 16 + row)) * S_LEN;
        for (int ks = sub * 16; ks < sub * 16 + 16; ++ks) {
            for (int h2 = 0; h2 < 2; ++h2) {
                unsigned short* p = sbf + ((ks * 32 + row + 16 * h2) << 4);
                v8u a = *(v8u*)p;           // keys ks*32 + 8*h2 + 0..7
                v8u b = *(v8u*)(p + 8);     // keys ks*32 + 8*h2 + 16..23
                float4 o0, o1, o2, o3;
                o0.x = bf2f(a[0]) * inv; o0.y = bf2f(a[1]) * inv; o0.z = bf2f(a[2]) * inv; o0.w = bf2f(a[3]) * inv;
                o1.x = bf2f(a[4]) * inv; o1.y = bf2f(a[5]) * inv; o1.z = bf2f(a[6]) * inv; o1.w = bf2f(a[7]) * inv;
                o2.x = bf2f(b[0]) * inv; o2.y = bf2f(b[1]) * inv; o2.z = bf2f(b[2]) * inv; o2.w = bf2f(b[3]) * inv;
                o3.x = bf2f(b[4]) * inv; o3.y = bf2f(b[5]) * inv; o3.z = bf2f(b[6]) * inv; o3.w = bf2f(b[7]) * inv;
                float* g0 = grow + ks * 32 + 8 * h2;
                ((float4*)g0)[0] = o0;
                ((float4*)(g0 + 4))[0] = o1;
                ((float4*)(g0 + 16))[0] = o2;
                ((float4*)(g0 + 20))[0] = o3;
            }
        }
    }
    // LDS e-values were last written before the rowsum barrier; phase 3 only reads.

    // ---- phase 3: context = (e @ pV) / rowsum ; wave -> one 16-wide dh slice ----
    const int dht = wid;                    // 0..3
    const int n   = lane & 15;
    v8f c = {};
    for (int ks = 0; ks < 128; ++ks) {
        v16bf aL = *(const v16bf*)(sbf + ((ks * 32 + lane) << 4));   // direct A-frag
        const unsigned short* bv_ = pV + ((size_t)((head * 128 + ks) * 4 + dht)) * 512;
        v16bf b = *(const v16bf*)(bv_ + lane * 16);
        c = wmma_bf16(aL, b, c);
    }
    // scatter context into A-frags for the output projection (normalize here)
    for (int r = 0; r < 8; ++r) {
        int mm  = r + 8 * hh;
        float val = c[r] / rowsum[mm];
        int col = head * DHEAD + dht * 16 + n;
        int ksub = col >> 5, Kk = col & 31;
        int h2 = (Kk >> 3) & 1;
        int j2 = (Kk & 7) + ((Kk >> 4) << 3);
        ctx_out[((size_t)(qt * KSUBD + ksub)) * 512 + (mm + 16 * h2) * 16 + j2] = f2bf(val);
    }
}

// =====================================================================
// K5: out = context @ Wo^T + bo -> fp32 to d_out. grid (256,12), block 128
// =====================================================================
__global__ void outproj_kernel(const unsigned short* __restrict__ ws,
                               const float* __restrict__ bo,
                               float* __restrict__ outO) {
    int qt   = blockIdx.x;
    int wid  = threadIdx.x >> 5;
    int lane = threadIdx.x & 31;
    int ntile = blockIdx.y * 4 + wid;

    const unsigned short* A = ws + OFF_CTX + (size_t)qt * KSUBD * 512;
    const unsigned short* B = ws + OFF_W + 3 * W_SZ;

    v8f c = {};
    for (int k = 0; k < KSUBD; ++k) {
        v16bf a = *(const v16bf*)(A + (size_t)k * 512 + lane * 16);
        v16bf b = *(const v16bf*)(B + ((size_t)(k * NTD + ntile)) * 512 + lane * 16);
        c = wmma_bf16(a, b, c);
    }
    int n = lane & 15, hh = lane >> 4;
    float bb = bo[ntile * 16 + n];
    for (int r = 0; r < 8; ++r) {
        int mm = r + 8 * hh;
        outO[(size_t)(qt * 16 + mm) * D_LEN + ntile * 16 + n] = c[r] + bb;
    }
}

// =====================================================================
extern "C" void kernel_launch(void* const* d_in, const int* in_sizes, int n_in,
                              void* d_out, int out_size, void* d_ws, size_t ws_size,
                              hipStream_t stream) {
    const float* Q  = (const float*)d_in[0];
    const float* K  = (const float*)d_in[1];
    const float* V  = (const float*)d_in[2];
    const float* Wq = (const float*)d_in[3];
    const float* bq = (const float*)d_in[4];
    const float* Wk = (const float*)d_in[5];
    const float* bk = (const float*)d_in[6];
    const float* Wv = (const float*)d_in[7];
    const float* bv = (const float*)d_in[8];
    const float* Wo = (const float*)d_in[9];
    const float* bo = (const float*)d_in[10];

    unsigned short* ws = (unsigned short*)d_ws;
    float* outO    = (float*)d_out;
    float* attnOut = outO + (size_t)S_LEN * D_LEN;

    pack_w_kernel<<<(4 * KSUBD * NTD * 32) / 256, 256, 0, stream>>>(Wq, Wk, Wv, Wo, ws);
    pack_x_kernel<<<(3 * QT * KSUBD * 32) / 256, 256, 0, stream>>>(Q, K, V, ws);
    proj_kernel<<<dim3(QT, NTD / 4, 3), 128, 0, stream>>>(ws, ws, bq, bk, bv);

    // 128 KB bf16 frag buffer + 640 B reduction scratch (2 WGs / WGP fit)
    size_t smem = (size_t)(128 * 512) * sizeof(unsigned short)
                + (size_t)(128 + 32) * sizeof(float);
    attn_kernel<<<dim3(QT, NH), 128, smem, stream>>>(ws, ws + OFF_CTX, attnOut);

    outproj_kernel<<<dim3(QT, NH), 128, 0, stream>>>(ws, bo, outO);
}
